// HybridDrapeModel_27127013442188
// MI455X (gfx1250) — matrix operand order repaired and checked by reading, not compiled, not run on hardware
//
#include <hip/hip_runtime.h>

typedef _Float16 half_t;
typedef __attribute__((ext_vector_type(16))) _Float16 v16h;
typedef __attribute__((ext_vector_type(8)))  _Float16 v8h;
typedef __attribute__((ext_vector_type(8)))  float    v8f;

#define LN_EPS 1e-5f
#define N_NODES 20000
#define N_EDGES 60000
#define BATCH   8
#define N_LAYERS 10

#define WMMA_F16(a, b, c) \
    __builtin_amdgcn_wmma_f32_16x16x32_f16(false, (a), false, (b), (short)0, (c), false, false)

// ---------------------------------------------------------------------------
// Fragment loaders (layouts per CDNA5 ISA 7.12.2)
// A fragment: row-major f16 row pointer already offset to kt*32 + 8*(lane>=16).
__device__ __forceinline__ v16h ldA(const half_t* p) {
    v8h lo = *(const v8h*)(p);
    v8h hi = *(const v8h*)(p + 16);
    v16h a;
#pragma unroll
    for (int i = 0; i < 8; ++i) { a[i] = lo[i]; a[i + 8] = hi[i]; }
    return a;
}

// B fragment from pre-packed tile: 32 contiguous bytes per lane.
__device__ __forceinline__ v16h ldB(const half_t* p) {
    v8h lo = *(const v8h*)(p);
    v8h hi = *(const v8h*)(p + 8);
    v16h b;
#pragma unroll
    for (int i = 0; i < 8; ++i) { b[i] = lo[i]; b[i + 8] = hi[i]; }
    return b;
}

// ---------------------------------------------------------------------------
// Weight packer: W (K x N, row-major f32, K may be < Kpad) -> fragment order.
// tile id = nt*nkt + kt ; per tile 512 f16 = [lane][16 elements].
// B element i of lane: k = kt*32 + (i&7) + 8*(lane>=16) + 16*(i>=8), n = nt*16 + lane%16.
// ---------------------------------------------------------------------------
__global__ __launch_bounds__(256) void k_pack(const float* __restrict__ W,
                                              half_t* __restrict__ dst,
                                              int K, int N, int nkt, int ntiles) {
    int tile = blockIdx.x * 8 + (threadIdx.x >> 5);
    int lane = threadIdx.x & 31;
    if (tile >= ntiles) return;
    int nt = tile / nkt, kt = tile % nkt;
    int hs = lane >> 4;
    int n  = nt * 16 + (lane & 15);
    half_t* d = dst + (size_t)tile * 512 + lane * 16;
#pragma unroll
    for (int i = 0; i < 16; ++i) {
        int k = kt * 32 + (i & 7) + (hs << 3) + ((i >> 3) << 4);
        float v = (k < K && n < N) ? W[(size_t)k * N + n] : 0.f;
        d[i] = (half_t)v;
    }
}

// ---------------------------------------------------------------------------
// Shared LN+ReLU epilogue over a 32x256 fp32 LDS tile -> f16 out.
// ---------------------------------------------------------------------------
__device__ __forceinline__ void ln_relu_epilogue(float* sAcc, float* sP, float* sP2,
                                                 float* sMean, float* sRstd,
                                                 const float* gam, const float* bet,
                                                 half_t* out, long rowBase) {
    const int tid = threadIdx.x;
    const int cg = tid & 15, rA = tid >> 4, rB = rA + 16;
    float sa = 0.f, sas = 0.f, sb = 0.f, sbs = 0.f;
#pragma unroll
    for (int j = 0; j < 16; ++j) {
        float va = sAcc[rA * 256 + cg * 16 + j]; sa += va; sas += va * va;
        float vb = sAcc[rB * 256 + cg * 16 + j]; sb += vb; sbs += vb * vb;
    }
    sP[rA * 16 + cg] = sa; sP2[rA * 16 + cg] = sas;
    sP[rB * 16 + cg] = sb; sP2[rB * 16 + cg] = sbs;
    __syncthreads();
    if (cg == 0) {
#pragma unroll 2
        for (int rp = 0; rp < 2; ++rp) {
            int row = rA + 16 * rp;
            float S = 0.f, SS = 0.f;
#pragma unroll
            for (int j = 0; j < 16; ++j) { S += sP[row * 16 + j]; SS += sP2[row * 16 + j]; }
            float mean = S * (1.f / 256.f);
            float var  = SS * (1.f / 256.f) - mean * mean;
            sMean[row] = mean; sRstd[row] = rsqrtf(var + LN_EPS);
        }
    }
    __syncthreads();
#pragma unroll 2
    for (int rp = 0; rp < 2; ++rp) {
        int row = rA + 16 * rp;
        float mean = sMean[row], rstd = sRstd[row];
        half_t* orow = out + (rowBase + row) * 256;
#pragma unroll
        for (int j = 0; j < 16; ++j) {
            int c = cg * 16 + j;
            float v = (sAcc[row * 256 + c] - mean) * rstd * gam[c] + bet[c];
            v = v > 0.f ? v : 0.f;
            orow[c] = (half_t)v;
        }
    }
}

// ---------------------------------------------------------------------------
// Fused GEMM (M x K @ K x 256) + bias + LayerNorm + ReLU -> f16 out.
// Block: 256 threads (8 waves), 32 rows x 256 cols. Each wave: 2 M-tiles x
// 2 N-tiles = 4 WMMA accumulators; next K-step A/B prefetched into registers
// before issuing current WMMAs (hides global load latency behind 4 WMMAs).
// Safe for in-place A==out (block reads only its own 32 rows, writes after).
// ---------------------------------------------------------------------------
__global__ __launch_bounds__(256) void k_gemm_ln(const half_t* __restrict__ A, long lda,
                                                 const half_t* __restrict__ Bp,
                                                 const float* __restrict__ bias,
                                                 const float* __restrict__ gam,
                                                 const float* __restrict__ bet,
                                                 half_t* __restrict__ out, int nkt) {
    __shared__ float sAcc[32 * 256];
    __shared__ float sP[512], sP2[512];
    __shared__ float sMean[32], sRstd[32];

    const int tid  = threadIdx.x;
    const int wave = tid >> 5, lane = tid & 31;
    const int hs   = lane >> 4, m = lane & 15;
    const long rowBase = (long)blockIdx.x * 32;
    const half_t* Arow0 = A + (rowBase + m) * lda + 8 * hs;
    const half_t* Arow1 = A + (rowBase + 16 + m) * lda + 8 * hs;
    const int nt0 = wave * 2, nt1 = nt0 + 1;
    const half_t* Bl0 = Bp + ((size_t)nt0 * nkt) * 512 + lane * 16;
    const half_t* Bl1 = Bp + ((size_t)nt1 * nkt) * 512 + lane * 16;

    v8f acc00 = {}, acc01 = {}, acc10 = {}, acc11 = {};
    v16h a0 = ldA(Arow0), a1 = ldA(Arow1);
    v16h b0 = ldB(Bl0),   b1 = ldB(Bl1);
    for (int kt = 0; kt < nkt; ++kt) {
        v16h na0 = a0, na1 = a1, nb0 = b0, nb1 = b1;
        if (kt + 1 < nkt) {
            na0 = ldA(Arow0 + (kt + 1) * 32);
            na1 = ldA(Arow1 + (kt + 1) * 32);
            nb0 = ldB(Bl0 + (size_t)(kt + 1) * 512);
            nb1 = ldB(Bl1 + (size_t)(kt + 1) * 512);
        }
        acc00 = WMMA_F16(a0, b0, acc00);
        acc01 = WMMA_F16(a0, b1, acc01);
        acc10 = WMMA_F16(a1, b0, acc10);
        acc11 = WMMA_F16(a1, b1, acc11);
        a0 = na0; a1 = na1; b0 = nb0; b1 = nb1;
    }
    const int c0 = nt0 * 16 + m, c1 = nt1 * 16 + m;
    const float bc0 = bias[c0], bc1 = bias[c1];
#pragma unroll
    for (int r = 0; r < 8; ++r) {
        int row0 = r + 8 * hs, row1 = row0 + 16;
        sAcc[row0 * 256 + c0] = acc00[r] + bc0;
        sAcc[row0 * 256 + c1] = acc01[r] + bc1;
        sAcc[row1 * 256 + c0] = acc10[r] + bc0;
        sAcc[row1 * 256 + c1] = acc11[r] + bc1;
    }
    __syncthreads();
    ln_relu_epilogue(sAcc, sP, sP2, sMean, sRstd, gam, bet, out, rowBase);
}

// ---------------------------------------------------------------------------
// Edge-concat variant: A row r (edge id) is concat(x[src[r]], x[dst[r]], e[r]),
// K = 384 (nkt = 12). 32-wide K tiles never straddle a 128-boundary, so each
// K-step resolves to a single base pointer per M-tile row.
// ---------------------------------------------------------------------------
__global__ __launch_bounds__(256) void k_gemm_ln_ecat(const half_t* __restrict__ xh,
                                                      const half_t* __restrict__ eh,
                                                      const int* __restrict__ srcI,
                                                      const int* __restrict__ dstI,
                                                      const half_t* __restrict__ Bp,
                                                      const float* __restrict__ bias,
                                                      const float* __restrict__ gam,
                                                      const float* __restrict__ bet,
                                                      half_t* __restrict__ out) {
    const int nkt = 12;
    __shared__ float sAcc[32 * 256];
    __shared__ float sP[512], sP2[512];
    __shared__ float sMean[32], sRstd[32];

    const int tid  = threadIdx.x;
    const int wave = tid >> 5, lane = tid & 31;
    const int hs   = lane >> 4, m = lane & 15;
    const long rowBase = (long)blockIdx.x * 32;
    const long r0 = rowBase + m, r1 = rowBase + 16 + m;
    const half_t* base0[3] = { xh + (size_t)srcI[r0] * 128,
                               xh + (size_t)dstI[r0] * 128,
                               eh + (size_t)r0 * 128 };
    const half_t* base1[3] = { xh + (size_t)srcI[r1] * 128,
                               xh + (size_t)dstI[r1] * 128,
                               eh + (size_t)r1 * 128 };
    const int nt0 = wave * 2, nt1 = nt0 + 1;
    const half_t* Bl0 = Bp + ((size_t)nt0 * nkt) * 512 + lane * 16;
    const half_t* Bl1 = Bp + ((size_t)nt1 * nkt) * 512 + lane * 16;

    v8f acc00 = {}, acc01 = {}, acc10 = {}, acc11 = {};
    v16h a0 = ldA(base0[0] + 8 * hs), a1 = ldA(base1[0] + 8 * hs);
    v16h b0 = ldB(Bl0),               b1 = ldB(Bl1);
#pragma unroll
    for (int kt = 0; kt < nkt; ++kt) {
        v16h na0 = a0, na1 = a1, nb0 = b0, nb1 = b1;
        if (kt + 1 < nkt) {
            int kn = (kt + 1) * 32;
            int seg = kn >> 7;          // 0: x[src], 1: x[dst], 2: e
            int ko  = (kn & 127) + 8 * hs;
            na0 = ldA(base0[seg] + ko);
            na1 = ldA(base1[seg] + ko);
            nb0 = ldB(Bl0 + (size_t)(kt + 1) * 512);
            nb1 = ldB(Bl1 + (size_t)(kt + 1) * 512);
        }
        acc00 = WMMA_F16(a0, b0, acc00);
        acc01 = WMMA_F16(a0, b1, acc01);
        acc10 = WMMA_F16(a1, b0, acc10);
        acc11 = WMMA_F16(a1, b1, acc11);
        a0 = na0; a1 = na1; b0 = nb0; b1 = nb1;
    }
    const int c0 = nt0 * 16 + m, c1 = nt1 * 16 + m;
    const float bc0 = bias[c0], bc1 = bias[c1];
#pragma unroll
    for (int r = 0; r < 8; ++r) {
        int row0 = r + 8 * hs, row1 = row0 + 16;
        sAcc[row0 * 256 + c0] = acc00[r] + bc0;
        sAcc[row0 * 256 + c1] = acc01[r] + bc1;
        sAcc[row1 * 256 + c0] = acc10[r] + bc0;
        sAcc[row1 * 256 + c1] = acc11[r] + bc1;
    }
    __syncthreads();
    ln_relu_epilogue(sAcc, sP, sP2, sMean, sRstd, gam, bet, out, rowBase);
}

// ---------------------------------------------------------------------------
// Final linear of each MLP: (M x 256 @ 256 x 128) + bias (+ f32 residual).
// Block: 32 rows x 128 cols; wave = 1 N-tile x 2 M-tiles (B reused twice).
// Writes fp32 master + f16 copy.
// ---------------------------------------------------------------------------
__global__ __launch_bounds__(256) void k_gemm_out(const half_t* __restrict__ A, long lda,
                                                  const half_t* __restrict__ Bp,
                                                  const float* __restrict__ bias,
                                                  const float* __restrict__ resid,
                                                  float* __restrict__ outF,
                                                  half_t* __restrict__ outH, int nkt) {
    const int tid  = threadIdx.x;
    const int wave = tid >> 5, lane = tid & 31;
    const int hs   = lane >> 4, m = lane & 15;
    const long rowBase = (long)blockIdx.x * 32;
    const half_t* Arow0 = A + (rowBase + m) * lda + 8 * hs;
    const half_t* Arow1 = A + (rowBase + 16 + m) * lda + 8 * hs;
    const half_t* Bl = Bp + ((size_t)wave * nkt) * 512 + lane * 16;

    v8f acc0 = {}, acc1 = {};
    v16h a0 = ldA(Arow0), a1 = ldA(Arow1), b = ldB(Bl);
    for (int kt = 0; kt < nkt; ++kt) {
        v16h na0 = a0, na1 = a1, nb = b;
        if (kt + 1 < nkt) {
            na0 = ldA(Arow0 + (kt + 1) * 32);
            na1 = ldA(Arow1 + (kt + 1) * 32);
            nb  = ldB(Bl + (size_t)(kt + 1) * 512);
        }
        acc0 = WMMA_F16(a0, b, acc0);
        acc1 = WMMA_F16(a1, b, acc1);
        a0 = na0; a1 = na1; b = nb;
    }
    int c = wave * 16 + m;
    float bc = bias[c];
#pragma unroll
    for (int r = 0; r < 8; ++r) {
        long o0 = (rowBase + r + 8 * hs) * 128 + c;
        long o1 = (rowBase + 16 + r + 8 * hs) * 128 + c;
        float v0 = acc0[r] + bc;
        float v1 = acc1[r] + bc;
        if (resid) { v0 += resid[o0]; v1 += resid[o1]; }
        outF[o0] = v0; outH[o0] = (half_t)v0;
        outF[o1] = v1; outH[o1] = (half_t)v1;
    }
}

// ---------------------------------------------------------------------------
// Style head (B=8): gelu(vit@W1+b1) -> LN -> @W2+b2 -> style; style@clsW+clsb.
// ---------------------------------------------------------------------------
__global__ __launch_bounds__(256) void k_style(const float* __restrict__ vit,
                                               const float* __restrict__ W1, const float* __restrict__ b1,
                                               const float* __restrict__ gam, const float* __restrict__ bet,
                                               const float* __restrict__ W2, const float* __restrict__ b2,
                                               const float* __restrict__ clsW, const float* __restrict__ clsb,
                                               float* __restrict__ style, float* __restrict__ logits) {
    __shared__ float sh[8 * 256];
    __shared__ float sst[8 * 128];
    __shared__ float sMean[8], sRstd[8];
    int tid = threadIdx.x;
    for (int rb = 0; rb < 8; ++rb) {
        float acc = b1[tid];
        for (int k = 0; k < 384; ++k) acc += vit[rb * 384 + k] * W1[k * 256 + tid];
        sh[rb * 256 + tid] = 0.5f * acc * (1.f + erff(acc * 0.70710678118f));
    }
    __syncthreads();
    if (tid < 8) {
        float s = 0.f, ss = 0.f;
        for (int c = 0; c < 256; ++c) { float v = sh[tid * 256 + c]; s += v; ss += v * v; }
        float mean = s * (1.f / 256.f);
        float var  = ss * (1.f / 256.f) - mean * mean;
        sMean[tid] = mean; sRstd[tid] = rsqrtf(var + LN_EPS);
    }
    __syncthreads();
    for (int rb = 0; rb < 8; ++rb)
        sh[rb * 256 + tid] = (sh[rb * 256 + tid] - sMean[rb]) * sRstd[rb] * gam[tid] + bet[tid];
    __syncthreads();
    if (tid < 128) {
        for (int rb = 0; rb < 8; ++rb) {
            float acc = b2[tid];
            for (int k = 0; k < 256; ++k) acc += sh[rb * 256 + k] * W2[k * 128 + tid];
            sst[rb * 128 + tid] = acc;
            style[rb * 128 + tid] = acc;
        }
    }
    __syncthreads();
    if (tid < 48) {
        int rb = tid / 6, c = tid % 6;
        float acc = clsb[c];
        for (int k = 0; k < 128; ++k) acc += sst[rb * 128 + k] * clsW[k * 6 + c];
        logits[rb * 6 + c] = acc;
    }
}

// ---------------------------------------------------------------------------
// Elementwise helpers
// ---------------------------------------------------------------------------
__global__ void k_feats(const float* __restrict__ pos, const float* __restrict__ uvs,
                        const float* __restrict__ nor, const float* __restrict__ style,
                        const float* __restrict__ smpl, const float* __restrict__ phys,
                        const float* __restrict__ tsz, const int* __restrict__ batch,
                        half_t* __restrict__ out, long total) {
    long idx = (long)blockIdx.x * blockDim.x + threadIdx.x;
    if (idx >= total) return;
    int n = (int)(idx / 160), c = (int)(idx % 160);
    int b = batch[n];
    float v;
    if      (c < 3)   v = pos[n * 3 + c];
    else if (c < 5)   v = uvs[n * 2 + (c - 3)];
    else if (c < 8)   v = nor[n * 3 + (c - 5)];
    else if (c < 136) v = style[b * 128 + (c - 8)];
    else if (c < 146) v = smpl[b * 10 + (c - 136)];
    else if (c < 156) v = phys[b * 10 + (c - 146)];
    else if (c < 158) v = tsz[b * 2 + (c - 156)];
    else              v = 0.f;
    out[idx] = (half_t)v;
}

__global__ void k_edgeattr(const float* __restrict__ ea, half_t* __restrict__ out, long total) {
    long idx = (long)blockIdx.x * blockDim.x + threadIdx.x;
    if (idx >= total) return;
    int e = (int)(idx >> 5), c = (int)(idx & 31);
    out[idx] = (half_t)(c < 4 ? ea[e * 4 + c] : 0.f);
}

__global__ void k_zero(float* __restrict__ p, long n) {
    long i = (long)blockIdx.x * blockDim.x + threadIdx.x;
    if (i < n) p[i] = 0.f;
}

__global__ void k_scatter(const float* __restrict__ e, const int* __restrict__ dstI,
                          float* __restrict__ agg, long total) {
    long idx = (long)blockIdx.x * blockDim.x + threadIdx.x;
    if (idx >= total) return;
    int ed = (int)(idx >> 7), c = (int)(idx & 127);
    atomicAdd(&agg[(size_t)dstI[ed] * 128 + c], e[idx]);
}

__global__ void k_nodein(const half_t* __restrict__ xh, const float* __restrict__ agg,
                         half_t* __restrict__ out, long total) {
    long idx = (long)blockIdx.x * blockDim.x + threadIdx.x;
    if (idx >= total) return;
    int n = (int)(idx >> 8), c = (int)(idx & 255);
    out[idx] = (c < 128) ? xh[(size_t)n * 128 + c] : (half_t)agg[(size_t)n * 128 + (c - 128)];
}

__global__ void k_decode(const float* __restrict__ x, const float* __restrict__ W,
                         const float* __restrict__ b, float* __restrict__ out, int N) {
    int n = blockIdx.x * blockDim.x + threadIdx.x;
    if (n >= N) return;
    float a0 = b[0], a1 = b[1], a2 = b[2];
    const float* xr = x + (size_t)n * 128;
    for (int k = 0; k < 128; ++k) {
        float v = xr[k];
        a0 += v * W[k * 3 + 0];
        a1 += v * W[k * 3 + 1];
        a2 += v * W[k * 3 + 2];
    }
    out[n * 3 + 0] = a0; out[n * 3 + 1] = a1; out[n * 3 + 2] = a2;
}

// ---------------------------------------------------------------------------
// Host orchestration
// ---------------------------------------------------------------------------
struct MlpP { const float *W1, *W2, *W3, *b1, *b2, *b3, *be1, *be2, *g1, *g2; };
struct MlpW { const half_t *W1p, *W2p, *W3p; };

extern "C" void kernel_launch(void* const* d_in, const int* in_sizes, int n_in,
                              void* d_out, int out_size, void* d_ws, size_t ws_size,
                              hipStream_t stream) {
    auto F = [&](int i) { return (const float*)d_in[i]; };

    const float* vit   = F(0);
    const float* pos   = F(1);
    const float* uvs   = F(2);
    const float* nor   = F(3);
    const float* eattr = F(4);
    const float* smpl  = F(5);
    const float* phys  = F(6);
    const float* tsz   = F(7);
    const int* eidx    = (const int*)d_in[8];
    const int* batch   = (const int*)d_in[9];
    const int* srcI = eidx;
    const int* dstI = eidx + N_EDGES;

    // params in JAX sorted-key pytree order
    const float* cls_W = F(10);
    const float* cls_b = F(11);
    const float* dec_W = F(12);
    const float* dec_b = F(13);
    auto mlpAt = [&](int base) {
        MlpP p; // sorted: W1,W2,W3,b1,b2,b3,be1,be2,g1,g2
        p.W1 = F(base + 0); p.W2 = F(base + 1); p.W3 = F(base + 2);
        p.b1 = F(base + 3); p.b2 = F(base + 4); p.b3 = F(base + 5);
        p.be1 = F(base + 6); p.be2 = F(base + 7); p.g1 = F(base + 8); p.g2 = F(base + 9);
        return p;
    };
    MlpP edge_enc = mlpAt(14);
    MlpP lyrE[N_LAYERS], lyrN[N_LAYERS];
    for (int i = 0; i < N_LAYERS; ++i) {
        lyrE[i] = mlpAt(24 + i * 20);
        lyrN[i] = mlpAt(24 + i * 20 + 10);
    }
    MlpP node_enc = mlpAt(224);
    const float* pW1 = F(234); const float* pW2 = F(235);
    const float* pb1 = F(236); const float* pb2 = F(237);
    const float* pbe = F(238); const float* pg  = F(239);

    // ---- workspace bump allocator ----
    char* wsb = (char*)d_ws;
    size_t cur = 0;
    auto alloc = [&](size_t bytes) { void* p = wsb + cur; cur = (cur + bytes + 255) & ~(size_t)255; return p; };
    float*  style = (float*) alloc(BATCH * 128 * sizeof(float));
    half_t* feats = (half_t*)alloc((size_t)N_NODES * 160 * 2);
    half_t* eah   = (half_t*)alloc((size_t)N_EDGES * 32 * 2);
    half_t* hid   = (half_t*)alloc((size_t)N_EDGES * 256 * 2);
    float*  xF    = (float*) alloc((size_t)N_NODES * 128 * 4);
    half_t* xH    = (half_t*)alloc((size_t)N_NODES * 128 * 2);
    float*  eF    = (float*) alloc((size_t)N_EDGES * 128 * 4);
    half_t* eH    = (half_t*)alloc((size_t)N_EDGES * 128 * 2);
    float*  agg   = (float*) alloc((size_t)N_NODES * 128 * 4);
    half_t* nin   = (half_t*)alloc((size_t)N_NODES * 256 * 2);
    half_t* wp    = (half_t*)alloc((size_t)4200000 * 2);
    half_t* wpc   = wp;

    auto packW = [&](const float* W, int K, int N, int Kpad) -> const half_t* {
        int nkt = Kpad / 32, ntiles = nkt * (N / 16);
        half_t* dst = wpc; wpc += (size_t)ntiles * 512;
        k_pack<<<(ntiles + 7) / 8, 256, 0, stream>>>(W, dst, K, N, nkt, ntiles);
        return dst;
    };

    // pack all GEMM weights (f32 -> f16 fragment order)
    MlpW neW { packW(node_enc.W1, 158, 256, 160), packW(node_enc.W2, 256, 256, 256), packW(node_enc.W3, 256, 128, 256) };
    MlpW eeW { packW(edge_enc.W1,   4, 256,  32), packW(edge_enc.W2, 256, 256, 256), packW(edge_enc.W3, 256, 128, 256) };
    MlpW lEW[N_LAYERS], lNW[N_LAYERS];
    for (int i = 0; i < N_LAYERS; ++i) {
        lEW[i] = MlpW{ packW(lyrE[i].W1, 384, 256, 384), packW(lyrE[i].W2, 256, 256, 256), packW(lyrE[i].W3, 256, 128, 256) };
        lNW[i] = MlpW{ packW(lyrN[i].W1, 256, 256, 256), packW(lyrN[i].W2, 256, 256, 256), packW(lyrN[i].W3, 256, 128, 256) };
    }

    float* out_delta  = (float*)d_out;               // (20000, 3)
    float* out_logits = (float*)d_out + N_NODES * 3; // (8, 6)

    // 1) style projection + classifier head
    k_style<<<1, 256, 0, stream>>>(vit, pW1, pb1, pg, pbe, pW2, pb2, cls_W, cls_b, style, out_logits);

    // 2) node features (N x 160 f16, padded from 158)
    {
        long tot = (long)N_NODES * 160;
        k_feats<<<(int)((tot + 255) / 256), 256, 0, stream>>>(pos, uvs, nor, style, smpl, phys, tsz, batch, feats, tot);
    }
    // 3) edge attr padded to K=32 f16
    {
        long tot = (long)N_EDGES * 32;
        k_edgeattr<<<(int)((tot + 255) / 256), 256, 0, stream>>>(eattr, eah, tot);
    }

    auto gemm_ln = [&](const half_t* A, long lda, const half_t* Bp, const MlpP& p, int stage,
                       half_t* out, int M, int nkt) {
        const float* b  = (stage == 1) ? p.b1 : p.b2;
        const float* g  = (stage == 1) ? p.g1 : p.g2;
        const float* be = (stage == 1) ? p.be1 : p.be2;
        k_gemm_ln<<<M / 32, 256, 0, stream>>>(A, lda, Bp, b, g, be, out, nkt);
    };
    auto gemm_out = [&](const half_t* A, const half_t* Bp, const float* b3, const float* resid,
                        float* oF, half_t* oH, int M) {
        k_gemm_out<<<M / 32, 256, 0, stream>>>(A, 256, Bp, b3, resid, oF, oH, 8);
    };

    // 4) node encoder: feats(160) -> 256 -> 256 -> x(128)
    gemm_ln(feats, 160, neW.W1p, node_enc, 1, hid, N_NODES, 5);
    gemm_ln(hid, 256, neW.W2p, node_enc, 2, hid, N_NODES, 8);
    gemm_out(hid, neW.W3p, node_enc.b3, nullptr, xF, xH, N_NODES);

    // 5) edge encoder: edge_attr(32) -> 256 -> 256 -> e(128)
    gemm_ln(eah, 32, eeW.W1p, edge_enc, 1, hid, N_EDGES, 1);
    gemm_ln(hid, 256, eeW.W2p, edge_enc, 2, hid, N_EDGES, 8);
    gemm_out(hid, eeW.W3p, edge_enc.b3, nullptr, eF, eH, N_EDGES);

    // 6) message passing layers
    for (int l = 0; l < N_LAYERS; ++l) {
        // e = e + edgeMLP(concat(x[src], x[dst], e))
        k_gemm_ln_ecat<<<N_EDGES / 32, 256, 0, stream>>>(xH, eH, srcI, dstI, lEW[l].W1p,
                                                         lyrE[l].b1, lyrE[l].g1, lyrE[l].be1, hid);
        gemm_ln(hid, 256, lEW[l].W2p, lyrE[l], 2, hid, N_EDGES, 8);
        gemm_out(hid, lEW[l].W3p, lyrE[l].b3, eF, eF, eH, N_EDGES);

        // agg = segment_sum(e, dst)
        {
            long nz = (long)N_NODES * 128;
            k_zero<<<(int)((nz + 255) / 256), 256, 0, stream>>>(agg, nz);
            long tot = (long)N_EDGES * 128;
            k_scatter<<<(int)((tot + 255) / 256), 256, 0, stream>>>(eF, dstI, agg, tot);
        }
        // x = x + nodeMLP(concat(x, agg))
        {
            long tot = (long)N_NODES * 256;
            k_nodein<<<(int)((tot + 255) / 256), 256, 0, stream>>>(xH, agg, nin, tot);
        }
        gemm_ln(nin, 256, lNW[l].W1p, lyrN[l], 1, hid, N_NODES, 8);
        gemm_ln(hid, 256, lNW[l].W2p, lyrN[l], 2, hid, N_NODES, 8);
        gemm_out(hid, lNW[l].W3p, lyrN[l].b3, xF, xF, xH, N_NODES);
    }

    // 7) decoder: predicted_delta = x @ dec_W + dec_b
    k_decode<<<(N_NODES + 255) / 256, 256, 0, stream>>>(xF, dec_W, dec_b, out_delta, N_NODES);

    (void)in_sizes; (void)n_in; (void)out_size; (void)ws_size;
}
